// Net_3040836845984
// MI455X (gfx1250) — matrix operand order repaired and checked by reading, not compiled, not run on hardware
//
#include <hip/hip_runtime.h>
#include <stdint.h>

// ---------------------------------------------------------------------------
// CDNA5 (gfx1250) types
// ---------------------------------------------------------------------------
typedef __attribute__((ext_vector_type(16))) __bf16 v16bf;
typedef __attribute__((ext_vector_type(8)))  float  v8f;
typedef unsigned int u32x4 __attribute__((ext_vector_type(4)));
typedef int          i32x8 __attribute__((ext_vector_type(8)));
typedef int          i32x4 __attribute__((ext_vector_type(4)));

__device__ __forceinline__ unsigned short f2bf(float f) {
  // round-to-nearest-even fp32 -> bf16
  unsigned int u = __float_as_uint(f);
  u += 0x7fffu + ((u >> 16) & 1u);
  return (unsigned short)(u >> 16);
}
__device__ __forceinline__ float bf2f(unsigned short h) {
  return __uint_as_float((unsigned int)h << 16);
}

// ---------------------------------------------------------------------------
// Tiled GEMM:  out[M,N](bf16) = X(bf16)[M,K] @ W(f32->bf16)[K,N] (+bias,relu)
//   block = 256 threads = 8 waves; tile = 128(M) x 64(N), K-step 32
//   A tile staged by the Tensor Data Mover (OOB rows/cols zero-filled by TDM),
//   W tile staged cooperatively (fp32 -> bf16, col-major for B fragments).
// ---------------------------------------------------------------------------
#define TM 128
#define TN 64
#define TK 32

__global__ __launch_bounds__(256) void gemm_bf16_wmma(
    const unsigned short* __restrict__ X, const float* __restrict__ W,
    const float* __restrict__ bias, unsigned short* __restrict__ out,
    int M, int K, int N, int relu_out)
{
  __shared__ __align__(16) unsigned short sX[TM * TK];  // [row][k]  row-major
  __shared__ __align__(16) unsigned short sW[TN * TK];  // [col][k]  col-major

  const int tid   = threadIdx.x;
  const int wave  = tid >> 5;
  const int lane  = tid & 31;
  const int hw    = lane >> 4;   // half-wave 0/1
  const int l16   = lane & 15;
  const int mbase = blockIdx.x * TM;
  const int nbase = blockIdx.y * TN;
  const unsigned int ldsOff = (unsigned int)(uintptr_t)&sX[0];  // LDS byte offset

  v8f acc[4] = {};

  for (int k0 = 0; k0 < K; k0 += TK) {
    __syncthreads();   // previous iteration's consumers are done with sX/sW

    // --- wave 0: issue TDM load of the 128x32 bf16 A tile into sX -----------
    if (wave == 0) {
      uint64_t ga = (uint64_t)(uintptr_t)(X + (size_t)mbase * K + k0);
      unsigned int dim0 = (unsigned int)(K - k0);   // remaining row elements
      unsigned int dim1 = (unsigned int)(M - mbase);// remaining rows
      u32x4 g0;
      g0[0] = 1u;                                   // count=1, is_restore=0
      g0[1] = ldsOff;                               // lds_addr (bytes)
      g0[2] = (unsigned int)ga;                     // global_addr[31:0]
      g0[3] = (unsigned int)((ga >> 32) & 0x01FFFFFFu) | (2u << 30); // addr[56:32] | type=2
      i32x8 g1;
      g1[0] = 0x00010000;                           // data_size=1 (2 bytes)
      g1[1] = (int)((dim0 & 0xFFFFu) << 16);        // tensor_dim0[15:0]
      g1[2] = (int)((dim0 >> 16) | ((dim1 & 0xFFFFu) << 16)); // dim0 hi | dim1 lo
      g1[3] = (int)((dim1 >> 16) | ((unsigned)TK << 16));     // dim1 hi | tile_dim0=32
      g1[4] = TM;                                   // tile_dim1=128, tile_dim2=0
      g1[5] = (int)(unsigned int)K;                 // tensor_dim0_stride (elements)
      g1[6] = 0;
      g1[7] = 0;
      i32x4 z4 = {0, 0, 0, 0};
#if __has_include(<hip/amd_detail/amd_gfx1250_TDM.h>)
      i32x8 z8 = {0, 0, 0, 0, 0, 0, 0, 0};
      __builtin_amdgcn_tensor_load_to_lds(g0, g1, z4, z4, z8, 0);
#else
      __builtin_amdgcn_tensor_load_to_lds(g0, g1, z4, z4, 0);
#endif
    }

    // --- all waves: stage W tile (fp32 -> bf16, col-major) ------------------
    for (int idx = tid; idx < TN * TK; idx += 256) {
      int c = idx >> 5, kk = idx & (TK - 1);
      int gk = k0 + kk;
      float v = (gk < K) ? W[(size_t)gk * N + (nbase + c)] : 0.f;
      sW[c * TK + kk] = f2bf(v);
    }
    // prefetch next W K-tile (emits global_prefetch_b8)
    if (k0 + TK < K) {
      int pk = k0 + TK + (tid >> 3);
      if (pk < K) __builtin_prefetch(&W[(size_t)pk * N + nbase + (tid & 7) * 8], 0, 1);
    }

    if (wave == 0) __builtin_amdgcn_s_wait_tensorcnt(0);
    __syncthreads();   // sX (TDM) + sW visible to all waves

    // A fragment (16x32 bf16, ISA 7.12.2): lanes 0-15 rows, K {0..7,16..23};
    // lanes 16-31 same rows, K {8..15,24..31}. Two 16B LDS chunks per lane.
    v16bf afrag;
    {
      const uint4* p = (const uint4*)&sX[(wave * 16 + l16) * TK + hw * 8];
      union { uint4 u[2]; v16bf v; } ua;
      ua.u[0] = p[0];   // K = hw*8 .. hw*8+7
      ua.u[1] = p[2];   // K = 16+hw*8 .. (+16 elem = +2 uint4)
      afrag = ua.v;
    }
    // B fragments (32x16 bf16): lane = col (mod 16), K {hw*8..hw*8+7, 16+hw*8..16+hw*8+7}
#pragma unroll
    for (int t = 0; t < 4; ++t) {
      const uint4* q = (const uint4*)&sW[(t * 16 + l16) * TK + hw * 8];
      union { uint4 u[2]; v16bf v; } ub;
      ub.u[0] = q[0];
      ub.u[1] = q[2];
      acc[t] = __builtin_amdgcn_wmma_f32_16x16x32_bf16(
          false, afrag, false, ub.v, (short)0, acc[t], false, false);
    }
  }

  // epilogue: C/D layout VGPR r: lanes 0-15 -> M=r, lanes 16-31 -> M=8+r
#pragma unroll
  for (int t = 0; t < 4; ++t) {
    int col = nbase + t * 16 + l16;
    float bv = bias ? bias[col] : 0.f;
#pragma unroll
    for (int r = 0; r < 8; ++r) {
      int row = mbase + wave * 16 + hw * 8 + r;
      if (row < M) {
        float v = acc[t][r] + bv;
        if (relu_out) v = fmaxf(v, 0.f);
        out[(size_t)row * N + col] = f2bf(v);
      }
    }
  }
}

// ---------------------------------------------------------------------------
// Graph / pooling / conversion kernels
// ---------------------------------------------------------------------------
__global__ void zero_kernel(float* p, int n) {
  int t = blockIdx.x * 256 + threadIdx.x;
  if (t < n) p[t] = 0.f;
}

// fp32 -> bf16 activation conversion with optional fused ReLU
__global__ void conv_kernel(const float* __restrict__ src, unsigned short* __restrict__ dst,
                            int n, int relu) {
  int t = blockIdx.x * 256 + threadIdx.x;
  if (t >= n) return;
  float v = src[t];
  if (relu) v = fmaxf(v, 0.f);
  dst[t] = f2bf(v);
}

__global__ void deg_kernel(const int* __restrict__ dst, float* __restrict__ deg, int E) {
  int e = blockIdx.x * 256 + threadIdx.x;
  if (e < E) atomicAdd(&deg[dst[e]], 1.0f);
}

__global__ void dinv_kernel(float* deg, int n) {
  int t = blockIdx.x * 256 + threadIdx.x;
  if (t < n) deg[t] = rsqrtf(1.0f + deg[t]);  // +1 self loop
}

// per-edge weight: dinv[src]*dinv[dst], computed once, reused by all channels
__global__ void edgew_kernel(const int* __restrict__ src, const int* __restrict__ dst,
                             const float* __restrict__ dinv, float* __restrict__ ew, int E) {
  int e = blockIdx.x * 256 + threadIdx.x;
  if (e < E) ew[e] = dinv[src[e]] * dinv[dst[e]];
}

// agg = h*dinv^2 + bias  (self-loop term, initializes scatter target), h in bf16
__global__ void agg_init_kernel(float* __restrict__ agg, const unsigned short* __restrict__ H,
                                const float* __restrict__ dinv,
                                const float* __restrict__ bias,
                                int total, int cbits) {
  int t = blockIdx.x * 256 + threadIdx.x;
  if (t >= total) return;
  int i = t >> cbits;
  int c = t & ((1 << cbits) - 1);
  float di = dinv[i];
  agg[t] = bf2f(H[t]) * di * di + bias[c];
}

// wave-per-edge scatter: agg[dst] += h[src]*w[e]; lane owns C/32 channels
__global__ __launch_bounds__(256) void scatter_kernel(
    float* __restrict__ agg, const unsigned short* __restrict__ H,
    const int* __restrict__ src, const int* __restrict__ dst,
    const float* __restrict__ ew, int E, int C) {
  int e = blockIdx.x * 8 + (threadIdx.x >> 5);
  if (e >= E) return;
  int lane = threadIdx.x & 31;
  int s = src[e], d = dst[e];
  float w = ew[e];
  int per = C >> 5;                       // 2 / 4 / 8 bf16 per lane
  const unsigned short* hp = H + (size_t)s * C + lane * per;
  float* ap = agg + (size_t)d * C + lane * per;
  for (int j = 0; j < per; ++j)
    atomicAdd(&ap[j], bf2f(hp[j]) * w);
}

__global__ void cnt_kernel(const int* __restrict__ batch, float* __restrict__ cnt, int n) {
  int t = blockIdx.x * 256 + threadIdx.x;
  if (t < n) atomicAdd(&cnt[batch[t]], 1.0f);
}

// mean pool with fused ReLU -> concatenated [512,512] FC input
__global__ void pool_kernel(const float* __restrict__ x, const int* __restrict__ batch,
                            float* __restrict__ pool, int total, int cbits, int colOff) {
  int t = blockIdx.x * 256 + threadIdx.x;
  if (t >= total) return;
  int i = t >> cbits;
  int c = t & ((1 << cbits) - 1);
  atomicAdd(&pool[(size_t)batch[i] * 512 + colOff + c], fmaxf(x[t], 0.f));
}

__global__ void pool_div_kernel(float* __restrict__ pool, const float* __restrict__ cnt,
                                int colOff) {
  int t = blockIdx.x * 256 + threadIdx.x;  // 512 graphs * 256 channels
  int g = t >> 8, c = t & 255;
  pool[(size_t)g * 512 + colOff + c] *= (1.0f / fmaxf(cnt[g], 1.0f));
}

// head: out[row] = relu(fc2)[row] . out_W + out_b   (fc2 already bf16+ReLU)
__global__ void out_kernel(const unsigned short* __restrict__ fc2,
                           const float* __restrict__ Wo,
                           const float* __restrict__ bo, float* __restrict__ out) {
  int row = blockIdx.x * 256 + threadIdx.x;
  if (row >= 512) return;
  float s = 0.f;
  for (int c = 0; c < 512; ++c) s += bf2f(fc2[row * 512 + c]) * Wo[c];
  out[row] = s + bo[0];
}

// ---------------------------------------------------------------------------
// Host-side orchestration
// ---------------------------------------------------------------------------
static inline int cdiv(int a, int b) { return (a + b - 1) / b; }

static void launch_gemm(const unsigned short* X, const float* W, const float* bias,
                        unsigned short* out, int M, int K, int N, int relu_out,
                        hipStream_t s) {
  dim3 g(cdiv(M, TM), N / TN);
  gemm_bf16_wmma<<<g, 256, 0, s>>>(X, W, bias, out, M, K, N, relu_out);
}

static void run_branch(const float* x, const int* ei, const int* batch,
                       int Nn, int E, int F,
                       const float* W1, const float* b1,
                       const float* W2, const float* b2,
                       const float* W3, const float* b3,
                       float* dinv, float* ew, unsigned short* xb, unsigned short* Hb,
                       float* aggA, float* aggB,
                       float* fcin, float* cnt, int colOff, hipStream_t s)
{
  const int* src = ei;
  const int* dst = ei + E;

  zero_kernel<<<cdiv(Nn, 256), 256, 0, s>>>(dinv, Nn);
  deg_kernel<<<cdiv(E, 256), 256, 0, s>>>(dst, dinv, E);
  dinv_kernel<<<cdiv(Nn, 256), 256, 0, s>>>(dinv, Nn);
  edgew_kernel<<<cdiv(E, 256), 256, 0, s>>>(src, dst, dinv, ew, E);

  // layer 1 (raw features; no ReLU)
  conv_kernel<<<cdiv(Nn * F, 256), 256, 0, s>>>(x, xb, Nn * F, 0);
  launch_gemm(xb, W1, nullptr, Hb, Nn, F, 64, 0, s);
  agg_init_kernel<<<cdiv(Nn * 64, 256), 256, 0, s>>>(aggA, Hb, dinv, b1, Nn * 64, 6);
  scatter_kernel<<<cdiv(E, 8), 256, 0, s>>>(aggA, Hb, src, dst, ew, E, 64);

  // layer 2 (ReLU fused into conversion)
  conv_kernel<<<cdiv(Nn * 64, 256), 256, 0, s>>>(aggA, xb, Nn * 64, 1);
  launch_gemm(xb, W2, nullptr, Hb, Nn, 64, 128, 0, s);
  agg_init_kernel<<<cdiv(Nn * 128, 256), 256, 0, s>>>(aggB, Hb, dinv, b2, Nn * 128, 7);
  scatter_kernel<<<cdiv(E, 8), 256, 0, s>>>(aggB, Hb, src, dst, ew, E, 128);

  // layer 3
  conv_kernel<<<cdiv(Nn * 128, 256), 256, 0, s>>>(aggB, xb, Nn * 128, 1);
  launch_gemm(xb, W3, nullptr, Hb, Nn, 128, 256, 0, s);
  agg_init_kernel<<<cdiv(Nn * 256, 256), 256, 0, s>>>(aggA, Hb, dinv, b3, Nn * 256, 8);
  scatter_kernel<<<cdiv(E, 8), 256, 0, s>>>(aggA, Hb, src, dst, ew, E, 256);

  // mean pool (ReLU fused) into fcin[:, colOff:colOff+256]
  zero_kernel<<<2, 256, 0, s>>>(cnt, 512);
  cnt_kernel<<<cdiv(Nn, 256), 256, 0, s>>>(batch, cnt, Nn);
  pool_kernel<<<cdiv(Nn * 256, 256), 256, 0, s>>>(aggA, batch, fcin, Nn * 256, 8, colOff);
  pool_div_kernel<<<cdiv(512 * 256, 256), 256, 0, s>>>(fcin, cnt, colOff);
}

extern "C" void kernel_launch(void* const* d_in, const int* in_sizes, int n_in,
                              void* d_out, int out_size, void* d_ws, size_t ws_size,
                              hipStream_t stream)
{
  (void)n_in; (void)out_size; (void)ws_size;

  const float* p_x  = (const float*)d_in[0];
  const int*   p_ei = (const int*)  d_in[1];
  const int*   p_bt = (const int*)  d_in[2];
  const float* l_x  = (const float*)d_in[3];
  const int*   l_ei = (const int*)  d_in[4];
  const int*   l_bt = (const int*)  d_in[5];
  const float* p_W1 = (const float*)d_in[6];  const float* p_b1 = (const float*)d_in[7];
  const float* p_W2 = (const float*)d_in[8];  const float* p_b2 = (const float*)d_in[9];
  const float* p_W3 = (const float*)d_in[10]; const float* p_b3 = (const float*)d_in[11];
  const float* l_W1 = (const float*)d_in[12]; const float* l_b1 = (const float*)d_in[13];
  const float* l_W2 = (const float*)d_in[14]; const float* l_b2 = (const float*)d_in[15];
  const float* l_W3 = (const float*)d_in[16]; const float* l_b3 = (const float*)d_in[17];
  const float* fc1W = (const float*)d_in[18]; const float* fc1b = (const float*)d_in[19];
  const float* fc2W = (const float*)d_in[20]; const float* fc2b = (const float*)d_in[21];
  const float* outW = (const float*)d_in[22]; const float* outb = (const float*)d_in[23];

  const int PN = in_sizes[2];              // 200000 protein nodes
  const int PE = in_sizes[1] / 2;          // 800000 protein edges
  const int PF = in_sizes[0] / PN;         // 41
  const int LN = in_sizes[5];              // 100000 ligand nodes
  const int LE = in_sizes[4] / 2;          // 400000 ligand edges
  const int LF = in_sizes[3] / LN;         // 78

  // workspace carve; branches run sequentially and share the large buffers
  char* ws = (char*)d_ws;
  size_t off = 0;
  auto carve = [&](size_t bytes) -> char* {
    char* p = ws + off;
    off += (bytes + 255) & ~(size_t)255;
    return p;
  };
  float*          dinvb = (float*)          carve((size_t)PN * 4);
  float*          ew    = (float*)          carve((size_t)PE * 4);
  unsigned short* xb    = (unsigned short*) carve((size_t)PN * 256 * 2);
  unsigned short* Hb    = (unsigned short*) carve((size_t)PN * 256 * 2);
  float*          aggA  = (float*)          carve((size_t)PN * 256 * 4);
  float*          aggB  = (float*)          carve((size_t)PN * 128 * 4);
  float*          fcin  = (float*)          carve((size_t)512 * 512 * 4);
  unsigned short* fcinb = (unsigned short*) carve((size_t)512 * 512 * 2);
  float*          cnt   = (float*)          carve((size_t)512 * 4);
  unsigned short* fc1o  = (unsigned short*) carve((size_t)512 * 1024 * 2);
  unsigned short* fc2o  = (unsigned short*) carve((size_t)512 * 512 * 2);

  zero_kernel<<<cdiv(512 * 512, 256), 256, 0, stream>>>(fcin, 512 * 512);

  run_branch(p_x, p_ei, p_bt, PN, PE, PF, p_W1, p_b1, p_W2, p_b2, p_W3, p_b3,
             dinvb, ew, xb, Hb, aggA, aggB, fcin, cnt, 0, stream);
  run_branch(l_x, l_ei, l_bt, LN, LE, LF, l_W1, l_b1, l_W2, l_b2, l_W3, l_b3,
             dinvb, ew, xb, Hb, aggA, aggB, fcin, cnt, 256, stream);

  // MLP head: bias + ReLU fused in bf16 epilogue, bf16 A feeds the next GEMM
  conv_kernel<<<cdiv(512 * 512, 256), 256, 0, stream>>>(fcin, fcinb, 512 * 512, 0);
  launch_gemm(fcinb, fc1W, fc1b, fc1o, 512, 512, 1024, 1, stream);
  launch_gemm(fc1o, fc2W, fc2b, fc2o, 512, 1024, 512, 1, stream);
  out_kernel<<<2, 256, 0, stream>>>(fc2o, outW, outb, (float*)d_out);
}
